// MultiHeadAttention_446676598965
// MI455X (gfx1250) — matrix operand order repaired and checked by reading, not compiled, not run on hardware
//
#include <hip/hip_runtime.h>

typedef __bf16 bf16_t;
typedef __attribute__((ext_vector_type(8)))  __bf16 v8bf;
typedef __attribute__((ext_vector_type(16))) __bf16 v16bf;
typedef __attribute__((ext_vector_type(8)))  float  v8f;
typedef __attribute__((ext_vector_type(4)))  float  v4f;

#define D_MODEL   1024
#define NUM_HEADS 16
#define D_HEAD    64
#define BATCH     4
#define SEQ       2048
#define ROWS      (BATCH * SEQ)   // 8192

static __device__ __forceinline__ bf16_t f2bf(float f) { return (bf16_t)f; }

// Concatenate two 8-element bf16 chunks into one 16-element WMMA fragment.
static __device__ __forceinline__ v16bf load_frag(const bf16_t* p0, const bf16_t* p1) {
    v8bf a = *(const v8bf*)p0;
    v8bf b = *(const v8bf*)p1;
    return __builtin_shufflevector(a, b, 0,1,2,3,4,5,6,7,8,9,10,11,12,13,14,15);
}

static __device__ __forceinline__ v8f wmma_bf16(v16bf a, v16bf b, v8f c) {
    return __builtin_amdgcn_wmma_f32_16x16x32_bf16(false, a, false, b, (short)0, c, false, false);
}

// ---------------- Stage 0: casts / transposes ----------------

__global__ void cast_x_kernel(const float* __restrict__ in, bf16_t* __restrict__ out, int n) {
    int i = (blockIdx.x * blockDim.x + threadIdx.x) * 4;
    if (i < n) {
        v4f v = *(const v4f*)(in + i);
        out[i + 0] = f2bf(v[0]);
        out[i + 1] = f2bf(v[1]);
        out[i + 2] = f2bf(v[2]);
        out[i + 3] = f2bf(v[3]);
    }
}

// out[n*K + k] = in[k*N + n]  (weights -> K-major bf16)
__global__ void transpose_cast_kernel(const float* __restrict__ in, bf16_t* __restrict__ out,
                                      int K, int N) {
    long long tid = (long long)blockIdx.x * blockDim.x + threadIdx.x;
    if (tid >= (long long)K * N) return;
    int k = (int)(tid % K);
    int n = (int)(tid / K);
    out[(long long)n * K + k] = f2bf(in[(long long)k * N + n]);
}

// ---------------- Stage 1: fused QKV projection ----------------
// A: x_bf16 [8192][1024]; Wt: [3072][1024] (K-major). 32x64 tile per wave.
// Q is pre-scaled by d_head^-0.5 at the epilogue; V stored transposed.

__global__ void __launch_bounds__(256)
qkv_gemm_kernel(const bf16_t* __restrict__ A, const bf16_t* __restrict__ Wt,
                const float* __restrict__ bias,
                bf16_t* __restrict__ Qo, bf16_t* __restrict__ Ko, bf16_t* __restrict__ Vt) {
    const int lane   = threadIdx.x & 31;
    const int lane15 = lane & 15;
    const int hi     = lane >> 4;
    const int wid    = blockIdx.x * 8 + (threadIdx.x >> 5);
    const int NG     = (3 * D_MODEL) / 64;       // 48 n-groups
    const int mt = wid / NG, ng = wid % NG;
    const int m0 = mt * 32, n0 = ng * 64;

    const int akoff = hi * 16;
    const int bkoff = hi * 16;

    v8f zero = {0,0,0,0,0,0,0,0};
    v8f acc[2][4];
#pragma unroll
    for (int mi = 0; mi < 2; ++mi)
#pragma unroll
        for (int t = 0; t < 4; ++t) acc[mi][t] = zero;

    const bf16_t* arow0 = A + (long long)(m0 + lane15) * D_MODEL;
    const bf16_t* arow1 = A + (long long)(m0 + 16 + lane15) * D_MODEL;

    for (int kb = 0; kb < D_MODEL; kb += 32) {
        // batch all fragment loads for this k-step, then burst the WMMAs
        v16bf aF[2];
        aF[0] = load_frag(arow0 + kb + akoff, arow0 + kb + 8 + akoff);
        aF[1] = load_frag(arow1 + kb + akoff, arow1 + kb + 8 + akoff);
        v16bf bF[4];
#pragma unroll
        for (int t = 0; t < 4; ++t) {
            const bf16_t* brow = Wt + (long long)(n0 + t * 16 + lane15) * D_MODEL + kb + bkoff;
            bF[t] = load_frag(brow, brow + 8);
        }
        if (kb + 128 < D_MODEL) {
            __builtin_prefetch(arow0 + kb + 128, 0, 0);
            __builtin_prefetch(arow1 + kb + 128, 0, 0);
        }
#pragma unroll
        for (int mi = 0; mi < 2; ++mi)
#pragma unroll
            for (int t = 0; t < 4; ++t)
                acc[mi][t] = wmma_bf16(aF[mi], bF[t], acc[mi][t]);
    }

#pragma unroll
    for (int t = 0; t < 4; ++t) {
        int n  = n0 + t * 16 + lane15;
        float bn = bias[n];
        int seg = n >> 10;          // 0=Q 1=K 2=V
        int nn  = n & 1023;
        int h   = nn >> 6, d = nn & 63;
#pragma unroll
        for (int mi = 0; mi < 2; ++mi) {
#pragma unroll
            for (int r = 0; r < 8; ++r) {
                int m = m0 + mi * 16 + r + hi * 8;
                int b = m >> 11, s = m & (SEQ - 1);
                float val = acc[mi][t][r] + bn;
                long long bh = (long long)(b * NUM_HEADS + h);
                if (seg == 0)      Qo[(bh * SEQ + s) * D_HEAD + d] = f2bf(val * 0.125f);
                else if (seg == 1) Ko[(bh * SEQ + s) * D_HEAD + d] = f2bf(val);
                else               Vt[(bh * D_HEAD + d) * SEQ + s] = f2bf(val);
            }
        }
    }
}

// ---------------- Stage 2: causal flash attention ----------------
// One wave per (batch*head, 16-row q tile). Online softmax, P via LDS transpose.
// Q already carries the 1/sqrt(d_head) scale.

__global__ void __launch_bounds__(256)
attn_kernel(const bf16_t* __restrict__ Q, const bf16_t* __restrict__ K,
            const bf16_t* __restrict__ Vt, bf16_t* __restrict__ O) {
    __shared__ bf16_t smem[8 * 16 * 32];   // 1KB per wave, 8 waves/block

    const int lane   = threadIdx.x & 31;
    const int lane15 = lane & 15;
    const int hi     = lane >> 4;
    const int wslot  = threadIdx.x >> 5;
    const int wid    = blockIdx.x * 8 + wslot;
    const int bh = wid >> 7;               // 0..63
    const int qt = wid & 127;
    const int q0 = qt * 16;
    const int b  = bh >> 4, h = bh & 15;

    const bf16_t* Qb = Q  + (long long)bh * SEQ * D_HEAD;
    const bf16_t* Kb = K  + (long long)bh * SEQ * D_HEAD;
    const bf16_t* Vb = Vt + (long long)bh * D_HEAD * SEQ;

    const int akoff = hi * 16;
    const int bkoff = hi * 16;

    v16bf qf[2];
#pragma unroll
    for (int ks = 0; ks < 2; ++ks) {
        const bf16_t* p = Qb + (q0 + lane15) * D_HEAD + ks * 32 + akoff;
        qf[ks] = load_frag(p, p + 8);
    }

    v8f zero = {0,0,0,0,0,0,0,0};
    v8f oacc[4];
#pragma unroll
    for (int t = 0; t < 4; ++t) oacc[t] = zero;
    float mrow[8], lrow[8];
#pragma unroll
    for (int r = 0; r < 8; ++r) { mrow[r] = -__builtin_inff(); lrow[r] = 0.f; }

    bf16_t* pls = smem + wslot * (16 * 32);

    for (int k0 = 0; k0 <= q0; k0 += 32) {
        // ---- QK^T: batch 4 K-fragments, then 4 WMMAs ----
        v16bf kf[4];   // [tile0 ks0, tile0 ks1, tile1 ks0, tile1 ks1]
#pragma unroll
        for (int ks = 0; ks < 2; ++ks) {
            const bf16_t* pb0 = Kb + (k0 + lane15) * D_HEAD + ks * 32 + bkoff;
            kf[ks] = load_frag(pb0, pb0 + 8);
            const bf16_t* pb1 = Kb + (k0 + 16 + lane15) * D_HEAD + ks * 32 + bkoff;
            kf[2 + ks] = load_frag(pb1, pb1 + 8);
        }
        if (k0 + 32 <= q0) {
            __builtin_prefetch(Kb + (k0 + 32 + lane15) * D_HEAD, 0, 0);
            __builtin_prefetch(Vb + lane15 * SEQ + k0 + 32, 0, 0);
        }
        v8f d0 = zero, d1 = zero;
        d0 = wmma_bf16(qf[0], kf[0], d0);
        d0 = wmma_bf16(qf[1], kf[1], d0);
        d1 = wmma_bf16(qf[0], kf[2], d1);
        d1 = wmma_bf16(qf[1], kf[3], d1);

        // ---- online softmax (rows live per lane-half; reduce across 16 lanes) ----
        const bool diag = (k0 + 32 > q0);
        float p0v[8], p1v[8];
#pragma unroll
        for (int r = 0; r < 8; ++r) {
            float x0 = d0[r], x1 = d1[r];
            if (diag) {
                int row = q0 + r + hi * 8;
                if (k0 + lane15 > row)      x0 = -__builtin_inff();
                if (k0 + 16 + lane15 > row) x1 = -__builtin_inff();
            }
            float mx = fmaxf(x0, x1);
#pragma unroll
            for (int off = 1; off < 16; off <<= 1) mx = fmaxf(mx, __shfl_xor(mx, off, 16));
            float mnew = fmaxf(mrow[r], mx);
            float corr = __expf(mrow[r] - mnew);
            float e0 = __expf(x0 - mnew), e1 = __expf(x1 - mnew);
            float rs = e0 + e1;
#pragma unroll
            for (int off = 1; off < 16; off <<= 1) rs += __shfl_xor(rs, off, 16);
            lrow[r] = lrow[r] * corr + rs;
            mrow[r] = mnew;
            p0v[r] = e0; p1v[r] = e1;
#pragma unroll
            for (int t = 0; t < 4; ++t) oacc[t][r] *= corr;
        }

        // ---- C-layout -> A-layout via LDS (bf16 16x32 tile, private per wave) ----
#pragma unroll
        for (int r = 0; r < 8; ++r) {
            int row = r + hi * 8;
            pls[row * 32 + lane15]      = f2bf(p0v[r]);
            pls[row * 32 + 16 + lane15] = f2bf(p1v[r]);
        }
        asm volatile("s_wait_dscnt 0" ::: "memory");
        const bf16_t* pp = pls + lane15 * 32 + akoff;
        v16bf pf = load_frag(pp, pp + 8);

        // ---- P*V: batch 4 V-fragments, then 4 WMMAs ----
        v16bf vf[4];
#pragma unroll
        for (int t = 0; t < 4; ++t) {
            const bf16_t* pv = Vb + (t * 16 + lane15) * SEQ + k0 + bkoff;
            vf[t] = load_frag(pv, pv + 8);
        }
#pragma unroll
        for (int t = 0; t < 4; ++t)
            oacc[t] = wmma_bf16(pf, vf[t], oacc[t]);
    }

    // normalize + store (bf16 activation for the out-proj GEMM)
#pragma unroll
    for (int r = 0; r < 8; ++r) {
        float inv = 1.0f / lrow[r];
        int s = q0 + r + hi * 8;
        long long rowbase = ((long long)b * SEQ + s) * D_MODEL + h * D_HEAD;
#pragma unroll
        for (int t = 0; t < 4; ++t)
            O[rowbase + t * 16 + lane15] = f2bf(oacc[t][r] * inv);
    }
}

// ---------------- Stage 3: output projection (f32 out + bias) ----------------
// 32x64 tile per wave, batched fragment loads.

__global__ void __launch_bounds__(256)
out_gemm_kernel(const bf16_t* __restrict__ A, const bf16_t* __restrict__ Wt,
                const float* __restrict__ bias, float* __restrict__ Out) {
    const int lane   = threadIdx.x & 31;
    const int lane15 = lane & 15;
    const int hi     = lane >> 4;
    const int wid    = blockIdx.x * 8 + (threadIdx.x >> 5);
    const int NG     = D_MODEL / 64;          // 16
    const int mt = wid / NG, ng = wid % NG;
    const int m0 = mt * 32, n0 = ng * 64;

    const int akoff = hi * 16;
    const int bkoff = hi * 16;

    v8f zero = {0,0,0,0,0,0,0,0};
    v8f acc[2][4];
#pragma unroll
    for (int mi = 0; mi < 2; ++mi)
#pragma unroll
        for (int t = 0; t < 4; ++t) acc[mi][t] = zero;

    const bf16_t* arow0 = A + (long long)(m0 + lane15) * D_MODEL;
    const bf16_t* arow1 = A + (long long)(m0 + 16 + lane15) * D_MODEL;

    for (int kb = 0; kb < D_MODEL; kb += 32) {
        v16bf aF[2];
        aF[0] = load_frag(arow0 + kb + akoff, arow0 + kb + 8 + akoff);
        aF[1] = load_frag(arow1 + kb + akoff, arow1 + kb + 8 + akoff);
        v16bf bF[4];
#pragma unroll
        for (int t = 0; t < 4; ++t) {
            const bf16_t* brow = Wt + (long long)(n0 + t * 16 + lane15) * D_MODEL + kb + bkoff;
            bF[t] = load_frag(brow, brow + 8);
        }
        if (kb + 128 < D_MODEL) {
            __builtin_prefetch(arow0 + kb + 128, 0, 0);
            __builtin_prefetch(arow1 + kb + 128, 0, 0);
        }
#pragma unroll
        for (int mi = 0; mi < 2; ++mi)
#pragma unroll
            for (int t = 0; t < 4; ++t)
                acc[mi][t] = wmma_bf16(aF[mi], bF[t], acc[mi][t]);
    }

#pragma unroll
    for (int t = 0; t < 4; ++t) {
        int n = n0 + t * 16 + lane15;
        float bn = bias[n];
#pragma unroll
        for (int mi = 0; mi < 2; ++mi) {
#pragma unroll
            for (int r = 0; r < 8; ++r) {
                int m = m0 + mi * 16 + r + hi * 8;
                Out[(long long)m * D_MODEL + n] = acc[mi][t][r] + bn;
            }
        }
    }
}

// ---------------- host ----------------

extern "C" void kernel_launch(void* const* d_in, const int* in_sizes, int n_in,
                              void* d_out, int out_size, void* d_ws, size_t ws_size,
                              hipStream_t stream) {
    const float* x     = (const float*)d_in[0];
    const float* w_qkv = (const float*)d_in[1];
    const float* b_qkv = (const float*)d_in[2];
    const float* w_out = (const float*)d_in[3];
    const float* b_out = (const float*)d_in[4];
    float* out = (float*)d_out;

    char* ws = (char*)d_ws;
    bf16_t* xb    = (bf16_t*)(ws + (0ll  << 20));   // 16 MiB: x in bf16
    bf16_t* wqkvT = (bf16_t*)(ws + (16ll << 20));   //  6 MiB: W_qkv^T bf16 [3072][1024]
    bf16_t* woutT = (bf16_t*)(ws + (22ll << 20));   //  2 MiB: W_out^T bf16 [1024][1024]
    bf16_t* q     = (bf16_t*)(ws + (24ll << 20));   // 16 MiB: Q [bh][s][64] (pre-scaled)
    bf16_t* k     = (bf16_t*)(ws + (40ll << 20));   // 16 MiB: K [bh][s][64]
    bf16_t* vT    = (bf16_t*)(ws + (56ll << 20));   // 16 MiB: V^T [bh][64][s]
    bf16_t* attn  = (bf16_t*)(ws + (72ll << 20));   // 16 MiB: attention output bf16

    cast_x_kernel<<<8192, 256, 0, stream>>>(x, xb, ROWS * D_MODEL);
    transpose_cast_kernel<<<12288, 256, 0, stream>>>(w_qkv, wqkvT, D_MODEL, 3 * D_MODEL);
    transpose_cast_kernel<<<4096, 256, 0, stream>>>(w_out, woutT, D_MODEL, D_MODEL);

    // 256 m-tiles * 48 n-groups = 12288 waves -> 1536 blocks of 8 waves
    qkv_gemm_kernel<<<1536, 256, 0, stream>>>(xb, wqkvT, b_qkv, q, k, vT);

    // 64 heads * 128 q-tiles = 8192 waves -> 1024 blocks
    attn_kernel<<<1024, 256, 0, stream>>>(q, k, vT, attn);

    // 256 m-tiles * 16 n-groups = 4096 waves -> 512 blocks
    out_gemm_kernel<<<512, 256, 0, stream>>>(attn, woutT, b_out, out);
}
